// DualAttention_12953621365338
// MI455X (gfx1250) — compile-verified, matrix operand order checked by reading, not run on hardware
//
#include <hip/hip_runtime.h>

// ---------------------------------------------------------------------------
// Problem constants: B=4, C=512, C8=64, H=W=64, N=4096
// ---------------------------------------------------------------------------
#define BB 4
#define CC 512
#define C8V 64
#define NN 4096
#define BN_EPS 1e-5f

typedef __attribute__((ext_vector_type(16))) __bf16 bf16x16;
typedef __attribute__((ext_vector_type(8)))  __bf16 bf16x8;
typedef __attribute__((ext_vector_type(8)))  float  f32x8;

union FragBF16 { bf16x16 v; bf16x8 h[2]; };

__device__ __forceinline__ unsigned short f2bf(float f) {
  unsigned u = __float_as_uint(f);
  u += 0x7FFFu + ((u >> 16) & 1u);   // round-to-nearest-even
  return (unsigned short)(u >> 16);
}
__device__ __forceinline__ float bf2f(unsigned short h) {
  return __uint_as_float(((unsigned)h) << 16);
}

// ---------------------------------------------------------------------------
// Generic NT GEMM: C[m,n] = sum_k A[m,k]*B[n,k], A:[M,K] bf16, B:[N,K] bf16.
// Macro-tile 128(M) x (BIGN?128:64)(N), BK=32, double-buffered LDS.
//  BIGN=1: 8 waves as 2x4, wave slab 64x32 = 4x2 WMMA  (1.5 ds_b128 / wmma)
//  BIGN=0: 8 waves as 4x2, wave slab 32x32 = 2x2 WMMA  (N=64 projections)
// One barrier per k-step; next tile's global loads issued before the barrier.
// ---------------------------------------------------------------------------
template<int BF16OUT, int BIGN>
__global__ __launch_bounds__(256) void wmma_nt_gemm(
    const unsigned short* __restrict__ Ag,
    const unsigned short* __restrict__ Bg,
    void* __restrict__ Cg,
    int M, int Ncols, int K,
    long long sA, long long sB, long long sC)
{
  constexpr int BN    = BIGN ? 128 : 64;
  constexpr int WN    = BIGN ? 4 : 2;      // waves in N
  constexpr int IT    = BIGN ? 4 : 2;      // 16-row subtiles per wave (M)
  constexpr int JT    = 2;                 // 16-col subtiles per wave (N)

  // rows padded to 40 halves (80B): conflict-free b128 fragment reads
  __shared__ __align__(16) unsigned short lA[2][128 * 40];
  __shared__ __align__(16) unsigned short lB[2][BN * 40];

  const int tid  = threadIdx.x;
  const int lane = tid & 31;
  const int wave = tid >> 5;
  const int wm   = wave / WN;
  const int wn   = wave % WN;
  const int m0   = blockIdx.x * 128;
  const int n0   = blockIdx.y * BN;

  const unsigned short* A = Ag + (long long)blockIdx.z * sA;
  const unsigned short* B = Bg + (long long)blockIdx.z * sB;

  f32x8 acc[IT][JT];
  #pragma unroll
  for (int i = 0; i < IT; i++)
    #pragma unroll
    for (int j = 0; j < JT; j++)
      #pragma unroll
      for (int r = 0; r < 8; r++) acc[i][j][r] = 0.0f;

  const int grow = tid >> 2;          // 0..63
  const int gc8  = (tid & 3) * 8;     // element column (x8 halves = 16B)
  const int lra  = wm * (IT * 16) + (lane & 15);
  const int lrb  = wn * 32 + (lane & 15);
  const int kb   = (lane & 16) ? 8 : 0;   // per-ISA 16-bit operand lane K layout

  const unsigned short* Ar0 = A + (size_t)(m0 + grow)      * K + gc8;
  const unsigned short* Ar1 = A + (size_t)(m0 + grow + 64) * K + gc8;
  const unsigned short* Br0 = B + (size_t)(n0 + grow)      * K + gc8;
  const unsigned short* Br1 = B + (size_t)(n0 + grow + (BIGN ? 64 : 0)) * K + gc8;

  // ---- prologue: stage k=0 tile into buffer 0 ----
  {
    uint4 a0 = *(const uint4*)(Ar0);
    uint4 a1 = *(const uint4*)(Ar1);
    uint4 b0 = *(const uint4*)(Br0);
    uint4 b1{};
    if constexpr (BIGN) b1 = *(const uint4*)(Br1);
    *(uint4*)(&lA[0][(grow)      * 40 + gc8]) = a0;
    *(uint4*)(&lA[0][(grow + 64) * 40 + gc8]) = a1;
    *(uint4*)(&lB[0][(grow)      * 40 + gc8]) = b0;
    if constexpr (BIGN) *(uint4*)(&lB[0][(grow + 64) * 40 + gc8]) = b1;
  }

  int cur = 0;
  for (int k0 = 32; k0 < K; k0 += 32, cur ^= 1) {
    // issue next tile's global loads (overlap with barrier + math)
    uint4 a0 = *(const uint4*)(Ar0 + k0);
    uint4 a1 = *(const uint4*)(Ar1 + k0);
    uint4 b0 = *(const uint4*)(Br0 + k0);
    uint4 b1{};
    if constexpr (BIGN) b1 = *(const uint4*)(Br1 + k0);

    __syncthreads();   // stores of buf[cur] (prev iter) visible to all waves

    FragBF16 af[IT], bfg[JT];
    #pragma unroll
    for (int i = 0; i < IT; i++) {
      const unsigned short* p = &lA[cur][(lra + i * 16) * 40 + kb];
      af[i].h[0] = *(const bf16x8*)(p);
      af[i].h[1] = *(const bf16x8*)(p + 16);
    }
    #pragma unroll
    for (int j = 0; j < JT; j++) {
      const unsigned short* p = &lB[cur][(lrb + j * 16) * 40 + kb];
      bfg[j].h[0] = *(const bf16x8*)(p);
      bfg[j].h[1] = *(const bf16x8*)(p + 16);
    }
    #pragma unroll
    for (int i = 0; i < IT; i++)
      #pragma unroll
      for (int j = 0; j < JT; j++)
        acc[i][j] = __builtin_amdgcn_wmma_f32_16x16x32_bf16(
            false, af[i].v, false, bfg[j].v, (short)0, acc[i][j], false, false);

    // stage next tile into the other buffer (no barrier needed: distinct buf)
    const int nxt = cur ^ 1;
    *(uint4*)(&lA[nxt][(grow)      * 40 + gc8]) = a0;
    *(uint4*)(&lA[nxt][(grow + 64) * 40 + gc8]) = a1;
    *(uint4*)(&lB[nxt][(grow)      * 40 + gc8]) = b0;
    if constexpr (BIGN) *(uint4*)(&lB[nxt][(grow + 64) * 40 + gc8]) = b1;
  }

  // ---- last tile ----
  __syncthreads();
  {
    FragBF16 af[IT], bfg[JT];
    #pragma unroll
    for (int i = 0; i < IT; i++) {
      const unsigned short* p = &lA[cur][(lra + i * 16) * 40 + kb];
      af[i].h[0] = *(const bf16x8*)(p);
      af[i].h[1] = *(const bf16x8*)(p + 16);
    }
    #pragma unroll
    for (int j = 0; j < JT; j++) {
      const unsigned short* p = &lB[cur][(lrb + j * 16) * 40 + kb];
      bfg[j].h[0] = *(const bf16x8*)(p);
      bfg[j].h[1] = *(const bf16x8*)(p + 16);
    }
    #pragma unroll
    for (int i = 0; i < IT; i++)
      #pragma unroll
      for (int j = 0; j < JT; j++)
        acc[i][j] = __builtin_amdgcn_wmma_f32_16x16x32_bf16(
            false, af[i].v, false, bfg[j].v, (short)0, acc[i][j], false, false);
  }

  // C layout: element (vgpr r, lane L) -> m = r + 8*(L>=16), n = L&15
  const long long cb0 = (long long)blockIdx.z * sC;
  const int rb0 = m0 + wm * (IT * 16) + ((lane >> 4) << 3);
  const int cb  = n0 + wn * 32 + (lane & 15);
  #pragma unroll
  for (int i = 0; i < IT; i++)
    #pragma unroll
    for (int j = 0; j < JT; j++) {
      int rb = rb0 + i * 16;
      int cc = cb + j * 16;
      #pragma unroll
      for (int r = 0; r < 8; r++) {
        float v = acc[i][j][r];
        size_t idx = (size_t)(cb0 + (long long)(rb + r) * Ncols + cc);
        if (BF16OUT) ((unsigned short*)Cg)[idx] = f2bf(v);
        else         ((float*)Cg)[idx] = v;
      }
    }
}

// ---------------------------------------------------------------------------
// Elementwise / reshape / softmax / BN kernels
// ---------------------------------------------------------------------------
__global__ __launch_bounds__(256) void zero_f32(float* p, int n) {
  int i = blockIdx.x * 256 + threadIdx.x;
  if (i < n) p[i] = 0.0f;
}

__global__ __launch_bounds__(256) void convert_bf16(const float* __restrict__ in,
                                                    unsigned short* __restrict__ out, int n) {
  int i = blockIdx.x * 256 + threadIdx.x;
  if (i < n) out[i] = f2bf(in[i]);
}

// x[B,C,N] f32 -> x_bf [B,C,N] bf16 and xT_bf [B,N,C] bf16 (LDS transpose)
__global__ __launch_bounds__(256) void transpose_convert_x(
    const float* __restrict__ x, unsigned short* __restrict__ xbf,
    unsigned short* __restrict__ xT) {
  __shared__ float t[64][65];
  const int tid = threadIdx.x;
  const int b = blockIdx.z, c0 = blockIdx.y * 64, n0 = blockIdx.x * 64;
  const int col = tid & 63, rb = tid >> 6;
  #pragma unroll
  for (int r = 0; r < 16; r++) {
    int crow = r * 4 + rb;
    size_t gi = ((size_t)(b * CC + c0 + crow)) * NN + n0 + col;
    float f = x[gi];
    xbf[gi] = f2bf(f);
    t[crow][col] = f;
  }
  __syncthreads();
  #pragma unroll
  for (int r = 0; r < 16; r++) {
    int nrow = r * 4 + rb;
    xT[((size_t)(b * NN + n0 + nrow)) * CC + c0 + col] = f2bf(t[col][nrow]);
  }
}

// softmax over rows of length 4096 (position attention), bf16 output
__global__ __launch_bounds__(256) void softmax_rows4096(
    const float* __restrict__ e, unsigned short* __restrict__ attn) {
  __shared__ float red[256];
  const int tid = threadIdx.x;
  const size_t base = (size_t)blockIdx.x * NN;
  float v[16];
  float m = -3.402823466e+38f;
  #pragma unroll
  for (int i = 0; i < 16; i++) { v[i] = e[base + tid + i * 256]; m = fmaxf(m, v[i]); }
  red[tid] = m; __syncthreads();
  for (int s = 128; s > 0; s >>= 1) { if (tid < s) red[tid] = fmaxf(red[tid], red[tid + s]); __syncthreads(); }
  m = red[0]; __syncthreads();
  float sum = 0.0f;
  #pragma unroll
  for (int i = 0; i < 16; i++) { v[i] = __expf(v[i] - m); sum += v[i]; }
  red[tid] = sum; __syncthreads();
  for (int s = 128; s > 0; s >>= 1) { if (tid < s) red[tid] += red[tid + s]; __syncthreads(); }
  float inv = 1.0f / red[0];
  #pragma unroll
  for (int i = 0; i < 16; i++) attn[base + tid + i * 256] = f2bf(v[i] * inv);
}

// channel attention: softmax(rowmax - ce) == exp(rowmin - ce)/sum, rows of 512
__global__ __launch_bounds__(256) void softmax_rows512_maxsub(
    const float* __restrict__ ce, unsigned short* __restrict__ cattn) {
  __shared__ float red[256];
  const int tid = threadIdx.x;
  const size_t base = (size_t)blockIdx.x * CC;
  float v0 = ce[base + tid], v1 = ce[base + tid + 256];
  red[tid] = fminf(v0, v1); __syncthreads();
  for (int s = 128; s > 0; s >>= 1) { if (tid < s) red[tid] = fminf(red[tid], red[tid + s]); __syncthreads(); }
  float mn = red[0]; __syncthreads();
  float e0 = __expf(mn - v0), e1 = __expf(mn - v1);
  red[tid] = e0 + e1; __syncthreads();
  for (int s = 128; s > 0; s >>= 1) { if (tid < s) red[tid] += red[tid + s]; __syncthreads(); }
  float inv = 1.0f / red[0];
  cattn[base + tid]       = f2bf(e0 * inv);
  cattn[base + tid + 256] = f2bf(e1 * inv);
}

// sT = bf16(gpa*paT + gca*caT + 2*xT)   (all [B,N,C] layout)
__global__ __launch_bounds__(256) void combine_s(
    const float* __restrict__ paT, const float* __restrict__ caT,
    const unsigned short* __restrict__ xT, unsigned short* __restrict__ sT,
    const float* __restrict__ gpa, const float* __restrict__ gca) {
  size_t i = (size_t)blockIdx.x * 256 + threadIdx.x;
  float s = gpa[0] * paT[i] + gca[0] * caT[i] + 2.0f * bf2f(xT[i]);
  sT[i] = f2bf(s);
}

// BN batch stats over yT [B*N, C]: per-channel sum/sumsq via block-local regs
__global__ __launch_bounds__(256) void bn_partial(
    const float* __restrict__ yT, float* __restrict__ sum, float* __restrict__ sumsq) {
  const int tid = threadIdx.x;
  const size_t r0 = (size_t)blockIdx.x * 64;
  float s0 = 0, q0 = 0, s1 = 0, q1 = 0;
  for (int r = 0; r < 64; r++) {
    const float* row = yT + (r0 + r) * CC;
    float a = row[tid];       s0 += a; q0 += a * a;
    float b = row[tid + 256]; s1 += b; q1 += b * b;
  }
  atomicAdd(&sum[tid], s0);         atomicAdd(&sumsq[tid], q0);
  atomicAdd(&sum[tid + 256], s1);   atomicAdd(&sumsq[tid + 256], q1);
}

__global__ void bn_finalize(const float* __restrict__ sum, const float* __restrict__ sumsq,
                            const float* __restrict__ scale, const float* __restrict__ bias,
                            float* __restrict__ a, float* __restrict__ b) {
  int c = threadIdx.x;
  if (c < CC) {
    const float invn = 1.0f / (float)(BB * NN);
    float mean = sum[c] * invn;
    float var  = sumsq[c] * invn - mean * mean;
    float sc   = rsqrtf(var + BN_EPS) * scale[c];
    a[c] = sc;
    b[c] = bias[c] - mean * sc;
  }
}

// yT [B,N,C] -> out [B,C,H,W] with BN affine + ReLU (LDS transpose)
__global__ __launch_bounds__(256) void finalize_out(
    const float* __restrict__ yT, const float* __restrict__ a,
    const float* __restrict__ bcoef, float* __restrict__ out) {
  __shared__ float t[64][65];
  const int tid = threadIdx.x;
  const int b = blockIdx.z, c0 = blockIdx.y * 64, n0 = blockIdx.x * 64;
  const int col = tid & 63, rb = tid >> 6;
  #pragma unroll
  for (int r = 0; r < 16; r++) {
    int nrow = r * 4 + rb;
    t[col][nrow] = yT[((size_t)(b * NN + n0 + nrow)) * CC + c0 + col];
  }
  __syncthreads();
  #pragma unroll
  for (int r = 0; r < 16; r++) {
    int crow = r * 4 + rb;
    int ch = c0 + crow;
    float v = t[crow][col] * a[ch] + bcoef[ch];
    out[((size_t)(b * CC + ch)) * NN + n0 + col] = fmaxf(v, 0.0f);
  }
}

// ---------------------------------------------------------------------------
// Host-side orchestration
// ---------------------------------------------------------------------------
extern "C" void kernel_launch(void* const* d_in, const int* in_sizes, int n_in,
                              void* d_out, int out_size, void* d_ws, size_t ws_size,
                              hipStream_t stream) {
  (void)in_sizes; (void)n_in; (void)out_size; (void)ws_size;
  const float* x   = (const float*)d_in[0];
  const float* wq  = (const float*)d_in[1];
  const float* wk  = (const float*)d_in[2];
  const float* wv  = (const float*)d_in[3];
  const float* wf  = (const float*)d_in[4];
  const float* gpa = (const float*)d_in[5];
  const float* gca = (const float*)d_in[6];
  const float* bns = (const float*)d_in[7];
  const float* bnb = (const float*)d_in[8];
  float* out = (float*)d_out;

  char* w = (char*)d_ws;
  // workspace layout (bytes)
  const size_t o_xbf   = 0;                        // [B,C,N] bf16   16 MB
  const size_t o_xT    = o_xbf   + 16777216;       // [B,N,C] bf16   16 MB
  const size_t o_vbf   = o_xT    + 16777216;       // [B,C,N] bf16   16 MB
  const size_t o_qT    = o_vbf   + 16777216;       // [B,N,64] bf16   2 MB
  const size_t o_kT    = o_qT    + 2097152;        // [B,N,64] bf16   2 MB
  const size_t o_cattn = o_kT    + 2097152;        // [B,C,C] bf16    2 MB
  const size_t o_ce    = o_cattn + 2097152;        // [B,C,C] f32     4 MB
  const size_t o_sT    = o_ce    + 4194304;        // [B,N,C] bf16   16 MB
  const size_t o_wq    = o_sT    + 16777216;       // bf16 weights
  const size_t o_wk    = o_wq    + 65536;
  const size_t o_wv    = o_wk    + 65536;
  const size_t o_wf    = o_wv    + 524288;
  const size_t o_bn    = o_wf    + 524288;         // sum|sumsq|a|b  4x2KB
  const size_t o_attn  = o_bn    + 8192;           // [B,N,N] bf16  134 MB
  const size_t o_paT   = o_attn  + 134217728;      // [B,N,C] f32    33 MB
  const size_t o_caT   = o_paT   + 33554432;       // [B,N,C] f32    33 MB
  const size_t o_yT    = o_caT   + 33554432;       // [B,N,C] f32    33 MB
  const size_t o_en    = o_yT    + 33554432;       // [B,N,N] f32   268 MB

  unsigned short* x_bf   = (unsigned short*)(w + o_xbf);
  unsigned short* xT_bf  = (unsigned short*)(w + o_xT);
  unsigned short* v_bf   = (unsigned short*)(w + o_vbf);
  unsigned short* qT_bf  = (unsigned short*)(w + o_qT);
  unsigned short* kT_bf  = (unsigned short*)(w + o_kT);
  unsigned short* ca_bf  = (unsigned short*)(w + o_cattn);
  float*          ce_f   = (float*)(w + o_ce);
  unsigned short* sT_bf  = (unsigned short*)(w + o_sT);
  unsigned short* wq_bf  = (unsigned short*)(w + o_wq);
  unsigned short* wk_bf  = (unsigned short*)(w + o_wk);
  unsigned short* wv_bf  = (unsigned short*)(w + o_wv);
  unsigned short* wf_bf  = (unsigned short*)(w + o_wf);
  float*          bn_sum = (float*)(w + o_bn);            // 512
  float*          bn_sq  = bn_sum + 512;                  // 512
  float*          bn_a   = bn_sum + 1024;
  float*          bn_b   = bn_sum + 1536;
  unsigned short* attn   = (unsigned short*)(w + o_attn);
  float*          paT    = (float*)(w + o_paT);
  float*          caT    = (float*)(w + o_caT);
  float*          yT     = (float*)(w + o_yT);
  float*          energy = (float*)(w + o_en);

  const long long sXT = (long long)NN * CC;   // 4096*512
  const long long sQK = (long long)NN * C8V;  // 4096*64
  const long long sCN = (long long)CC * NN;   // 512*4096
  const long long sNn = (long long)NN * NN;   // 4096*4096
  const long long sCc = (long long)CC * CC;   // 512*512

  // 0) zero BN accumulators (per-call: graph-replay determinism)
  zero_f32<<<4, 256, 0, stream>>>(bn_sum, 1024);

  // 1) weights -> bf16
  convert_bf16<<<128,  256, 0, stream>>>(wq, wq_bf, C8V * CC);
  convert_bf16<<<128,  256, 0, stream>>>(wk, wk_bf, C8V * CC);
  convert_bf16<<<1024, 256, 0, stream>>>(wv, wv_bf, CC * CC);
  convert_bf16<<<1024, 256, 0, stream>>>(wf, wf_bf, CC * CC);

  // 2) x -> x_bf [B,C,N] and xT_bf [B,N,C]
  transpose_convert_x<<<dim3(NN/64, CC/64, BB), 256, 0, stream>>>(x, x_bf, xT_bf);

  // 3) qT[n,o] = sum_c xT[n,c]*wq[o,c]  (M=4096,N=64,K=512) -> bf16
  wmma_nt_gemm<1,0><<<dim3(32, 1, BB), 256, 0, stream>>>(xT_bf, wq_bf, qT_bf,
      NN, C8V, CC, sXT, 0, sQK);
  wmma_nt_gemm<1,0><<<dim3(32, 1, BB), 256, 0, stream>>>(xT_bf, wk_bf, kT_bf,
      NN, C8V, CC, sXT, 0, sQK);

  // 4) v[c,n] = sum_k wv[c,k]*xT[n,k]   (M=512,N=4096,K=512) -> bf16 [C,N]
  wmma_nt_gemm<1,1><<<dim3(4, 32, BB), 256, 0, stream>>>(wv_bf, xT_bf, v_bf,
      CC, NN, CC, 0, sXT, sCN);

  // 5) energy[n,m] = sum_o qT[n,o]*kT[m,o]  (M=N=4096,K=64) -> f32
  wmma_nt_gemm<0,1><<<dim3(32, 32, BB), 256, 0, stream>>>(qT_bf, kT_bf, energy,
      NN, NN, C8V, sQK, sQK, sNn);

  // 6) position softmax -> attn bf16
  softmax_rows4096<<<BB * NN, 256, 0, stream>>>(energy, attn);

  // 7) ce[c,d] = sum_n x[c,n]*x[d,n]   (M=N=512,K=4096) -> f32
  wmma_nt_gemm<0,1><<<dim3(4, 4, BB), 256, 0, stream>>>(x_bf, x_bf, ce_f,
      CC, CC, NN, sCN, sCN, sCc);

  // 8) channel softmax(max - ce) -> cattn bf16
  softmax_rows512_maxsub<<<BB * CC, 256, 0, stream>>>(ce_f, ca_bf);

  // 9) paT[n,c] = sum_m attn[n,m]*v[c,m]  (M=4096,N=512,K=4096) -> f32
  wmma_nt_gemm<0,1><<<dim3(32, 4, BB), 256, 0, stream>>>(attn, v_bf, paT,
      NN, CC, NN, sNn, sCN, sXT);

  // 10) caT[n,c] = sum_d xT[n,d]*cattn[c,d]  (M=4096,N=512,K=512) -> f32
  wmma_nt_gemm<0,1><<<dim3(32, 4, BB), 256, 0, stream>>>(xT_bf, ca_bf, caT,
      NN, CC, CC, sXT, sCc, sXT);

  // 11) sT = bf16(gpa*paT + gca*caT + 2*xT)
  combine_s<<<(unsigned)((size_t)BB * NN * CC / 256), 256, 0, stream>>>(
      paT, caT, xT_bf, sT_bf, gpa, gca);

  // 12) yT[n,o] = sum_c sT[n,c]*wf[o,c]  (M=4096,N=512,K=512) -> f32
  wmma_nt_gemm<0,1><<<dim3(32, 4, BB), 256, 0, stream>>>(sT_bf, wf_bf, yT,
      NN, CC, CC, sXT, 0, sXT);

  // 13) BN batch stats + affine fold
  bn_partial<<<(BB * NN) / 64, 256, 0, stream>>>(yT, bn_sum, bn_sq);
  bn_finalize<<<1, 512, 0, stream>>>(bn_sum, bn_sq, bns, bnb, bn_a, bn_b);

  // 14) transpose + BN + ReLU -> d_out [B,C,H,W]
  finalize_out<<<dim3(NN/64, CC/64, BB), 256, 0, stream>>>(yT, bn_a, bn_b, out);
}
